// MalConv_82085414961600
// MI455X (gfx1250) — matrix operand order repaired and checked by reading, not compiled, not run on hardware
//
#include <hip/hip_runtime.h>

typedef __attribute__((ext_vector_type(16))) _Float16 v16h;
typedef __attribute__((ext_vector_type(8)))  _Float16 v8h;
typedef __attribute__((ext_vector_type(8)))  float    v8f;

#define KER    500
#define CH     8
#define OUTC   128
#define JDIM   4000          // 500 * 8
#define KSTEPS 125           // 4000 / 32
#define NPATCH_PER_B 2000    // 1e6 / 500
#define EMB_ROWS 257

union AV { v16h v; v8h h[2]; };

// ---- prep: repack W[o][c][k] (f32) -> Wt[o][j = k*8 + c] (f16) ----
__global__ void pack_w_kernel(const float* __restrict__ W, _Float16* __restrict__ Wt) {
    int i = blockIdx.x * blockDim.x + threadIdx.x;
    if (i < OUTC * JDIM) {
        int o = i / JDIM, j = i - o * JDIM;
        int k = j >> 3, c = j & 7;
        Wt[i] = (_Float16)W[(o * CH + c) * KER + k];
    }
}

__global__ void zero_kernel(float* __restrict__ p, int n) {
    int i = blockIdx.x * blockDim.x + threadIdx.x;
    if (i < n) p[i] = 0.0f;
}

// ---- main fused kernel: gather-GEMM x2 + GLU + per-batch channel max ----
// Block = 8 waves = 4 M-tiles x 2 N-halves. Each wave: 16 patches x 64 channels
// x 2 gates -> 8 v8f accumulators (64 VGPRs), 125 k-steps, 8 WMMA/step.
__global__ __launch_bounds__(256) void malconv_gemm_kernel(
    const int* __restrict__ x,
    const float* __restrict__ emb,
    const _Float16* __restrict__ Wt1,
    const _Float16* __restrict__ Wt2,
    const float* __restrict__ b1,
    const float* __restrict__ b2,
    float* __restrict__ mbuf /* [8][128], int-bit atomic max, pre-zeroed */)
{
    __shared__ _Float16 embtab[EMB_ROWS * CH];      // 4112 B, f16 embedding table
    for (int i = threadIdx.x; i < EMB_ROWS * CH; i += blockDim.x)
        embtab[i] = (_Float16)emb[i];
    __syncthreads();

    const int wave     = threadIdx.x >> 5;
    const int lane     = threadIdx.x & 31;
    const int half_sel = lane >> 4;                 // 0: lanes 0-15, 1: lanes 16-31
    const int lrow     = lane & 15;

    const int tile  = blockIdx.x * 4 + (wave >> 1); // 0..999 M-tiles of 16 patches
    const int nhalf = wave & 1;                     // 0: ch 0-63, 1: ch 64-127
    const int m0    = tile * 16;
    const int batch = m0 / NPATCH_PER_B;            // tiles never cross batches

    // A side: per lane two byte positions per k-step: k = 4*ks + 2*half_sel, +1
    const int mrow = m0 + lrow;
    const int* xp  = x + mrow * KER + 2 * half_sel;

    // B side: lane holds column n = nhalf*64 + nt*16 + lrow, 16 contiguous j's
    const _Float16* w1row = Wt1 + (nhalf * 64 + lrow) * JDIM + half_sel * 16;
    const _Float16* w2row = Wt2 + (nhalf * 64 + lrow) * JDIM + half_sel * 16;

    const v8h* tab = (const v8h*)embtab;

    v8f z = {};
    v8f acc1[4], acc2[4];
#pragma unroll
    for (int nt = 0; nt < 4; nt++) { acc1[nt] = z; acc2[nt] = z; }

#pragma unroll 2
    for (int ks = 0; ks < KSTEPS; ks++) {
        // ---- issue ALL loads for this step first (forms one load clause) ----
        int i0 = xp[0];
        int i1 = xp[1];
        xp += 4;
        __builtin_prefetch((const void*)xp, 0, 0);          // next step's indices
        AV a;
        a.h[0] = tab[i0];                                   // ds_load_b128
        a.h[1] = tab[i1];
        AV bf1[4], bf2[4];
#pragma unroll
        for (int nt = 0; nt < 4; nt++) {
            const v8h* p1 = (const v8h*)(w1row + nt * 16 * JDIM);
            bf1[nt].h[0] = p1[0]; bf1[nt].h[1] = p1[1];     // 2x global_load_b128
            const v8h* p2 = (const v8h*)(w2row + nt * 16 * JDIM);
            bf2[nt].h[0] = p2[0]; bf2[nt].h[1] = p2[1];
        }
        // ---- then all 8 WMMAs drain the loads ----
#pragma unroll
        for (int nt = 0; nt < 4; nt++) {
            acc1[nt] = __builtin_amdgcn_wmma_f32_16x16x32_f16(
                false, a.v, false, bf1[nt].v, (short)0, acc1[nt], false, false);
            acc2[nt] = __builtin_amdgcn_wmma_f32_16x16x32_f16(
                false, a.v, false, bf2[nt].v, (short)0, acc2[nt], false, false);
        }
        w1row += 32; w2row += 32;
    }

    // Epilogue: bias + GLU + relu, reduce max over this tile's 16 patches.
    // C layout: lane holds column n; VGPR r = one of 8 M rows
    // (lanes 0-15: M=r, lanes 16-31: M=8+r) -> reduce r, then xor-16 shuffle.
#pragma unroll
    for (int nt = 0; nt < 4; nt++) {
        int n = nhalf * 64 + nt * 16 + lrow;
        float bb1 = b1[n], bb2 = b2[n];
        float hm = 0.0f;
#pragma unroll
        for (int r = 0; r < 8; r++) {
            float g1 = acc1[nt][r] + bb1;
            float g2 = acc2[nt][r] + bb2;
            float sg = 1.0f / (1.0f + __expf(-g2));
            float h  = g1 * sg;
            hm = fmaxf(hm, fmaxf(h, 0.0f));
        }
        hm = fmaxf(hm, __shfl_xor(hm, 16, 32));
        // relu output >= 0 -> IEEE order == int order on the bit pattern
        if (half_sel == 0)
            atomicMax((int*)(mbuf + batch * OUTC + n), __float_as_int(hm));
    }
}

// ---- final FC: out[b][cls] = mbuf[b] . fcW[cls] + fcb[cls] (16 outputs) ----
__global__ void fc_kernel(const float* __restrict__ mbuf, const float* __restrict__ fcW,
                          const float* __restrict__ fcb, float* __restrict__ out) {
    int t = threadIdx.x;
    if (t < 16) {
        int b = t >> 1, cls = t & 1;
        float s = fcb[cls];
        for (int o = 0; o < OUTC; o++)
            s += mbuf[b * OUTC + o] * fcW[cls * OUTC + o];
        out[t] = s;
    }
}

extern "C" void kernel_launch(void* const* d_in, const int* in_sizes, int n_in,
                              void* d_out, int out_size, void* d_ws, size_t ws_size,
                              hipStream_t stream) {
    const int*   x   = (const int*)d_in[0];
    const float* emb = (const float*)d_in[1];
    const float* W1  = (const float*)d_in[2];
    const float* b1  = (const float*)d_in[3];
    const float* W2  = (const float*)d_in[4];
    const float* b2  = (const float*)d_in[5];
    const float* fcW = (const float*)d_in[6];
    const float* fcb = (const float*)d_in[7];
    float* out = (float*)d_out;

    char* ws = (char*)d_ws;
    _Float16* Wt1 = (_Float16*)ws;                                  // 1,024,000 B
    _Float16* Wt2 = (_Float16*)(ws + (size_t)OUTC * JDIM * 2);      // 1,024,000 B
    float*    mbuf = (float*)(ws + 2 * (size_t)OUTC * JDIM * 2);    // 4,096 B

    const int nW = OUTC * JDIM;
    pack_w_kernel<<<(nW + 255) / 256, 256, 0, stream>>>(W1, Wt1);
    pack_w_kernel<<<(nW + 255) / 256, 256, 0, stream>>>(W2, Wt2);
    zero_kernel<<<(8 * OUTC + 255) / 256, 256, 0, stream>>>(mbuf, 8 * OUTC);
    malconv_gemm_kernel<<<250, 256, 0, stream>>>(x, emb, Wt1, Wt2, b1, b2, mbuf);
    fc_kernel<<<1, 64, 0, stream>>>(mbuf, fcW, fcb, out);
}
